// IDHead_53309134078123
// MI455X (gfx1250) — matrix operand order repaired and checked by reading, not compiled, not run on hardware
//
#include <hip/hip_runtime.h>
#include <hip/hip_bf16.h>
#include <stdint.h>

typedef __attribute__((ext_vector_type(16))) _Float16 v16h;
typedef __attribute__((ext_vector_type(8)))  float    v8f;

#define DDIM   512     // feature dim (fixed by reference)
#define NT     128     // bank columns (rows of lab_bank) per block/chunk
#define BSTR   520     // padded f16 row stride in LDS (bank-conflict dodge)
#define NWAVES 8       // 256 threads = 8 wave32
#define TOPK   5

union Frag16 { v16h v; uint32_t u[8]; };

// ---------------------------------------------------------------------------
// Kernel 1: row-normalize z (f32) -> f16, one wave per row.
// ---------------------------------------------------------------------------
__global__ void __launch_bounds__(256)
norm_rows_f16(const float* __restrict__ src, _Float16* __restrict__ dst, int rows) {
  const int lane = threadIdx.x & 31;
  const int wave = threadIdx.x >> 5;
  const int row  = blockIdx.x * NWAVES + wave;
  if (row >= rows) return;
  const float* s = src + (size_t)row * DDIM;
  float vals[16];
  float ss = 0.f;
#pragma unroll
  for (int i = 0; i < 16; ++i) {
    float v = s[i * 32 + lane];
    vals[i] = v;
    ss += v * v;
  }
#pragma unroll
  for (int off = 16; off > 0; off >>= 1) ss += __shfl_xor(ss, off, 32);
  const float scale = 1.0f / fmaxf(sqrtf(ss), 1e-12f);
  _Float16* d = dst + (size_t)row * DDIM;
#pragma unroll
  for (int i = 0; i < 16; ++i) d[i * 32 + lane] = (_Float16)(vals[i] * scale);
}

// ---------------------------------------------------------------------------
// Kernel 2: per-chunk cosine sims via WMMA f16 + chunk-local top-5.
// Block = one chunk of NT bank rows. 8 waves sweep all B/16 M-tiles of zn.
// ---------------------------------------------------------------------------
__global__ void __launch_bounds__(256)
simtopk_kernel(const _Float16* __restrict__ zn, const float* __restrict__ bank,
               float* __restrict__ cand_val, int* __restrict__ cand_idx,
               int N, int B) {
  __shared__ __align__(16) _Float16 tileB[NT * BSTR];          // ~130 KB
  __shared__ __align__(16) float    simBuf[NWAVES * 16 * NT];  // 64 KB

  const int lane = threadIdx.x & 31;
  const int wave = threadIdx.x >> 5;
  const int hf   = lane >> 4;    // half-wave: 0 = lanes 0-15, 1 = lanes 16-31
  const int sub  = lane & 15;
  const int chunkBase = blockIdx.x * NT;

  // ---- Phase A: stream NT bank rows from HBM, normalize, store f16 in LDS.
  for (int rr = 0; rr < NT / NWAVES; ++rr) {
    const int localRow = wave * (NT / NWAVES) + rr;
    const int grow = chunkBase + localRow;
    _Float16* drow = tileB + localRow * BSTR;
    if (grow < N) {
      const float* s = bank + (size_t)grow * DDIM;
      if (grow + NT / NWAVES < N)  // lookahead prefetch of the next row group
        __builtin_prefetch(bank + (size_t)(grow + NT / NWAVES) * DDIM + lane * 16, 0, 1);
      float vals[16];
      float ss = 0.f;
#pragma unroll
      for (int i = 0; i < 16; ++i) {
        float v = s[i * 32 + lane];
        vals[i] = v;
        ss += v * v;
      }
#pragma unroll
      for (int off = 16; off > 0; off >>= 1) ss += __shfl_xor(ss, off, 32);
      const float scale = 1.0f / fmaxf(sqrtf(ss), 1e-12f);
#pragma unroll
      for (int i = 0; i < 16; ++i) drow[i * 32 + lane] = (_Float16)(vals[i] * scale);
    } else {
#pragma unroll
      for (int i = 0; i < 16; ++i) drow[i * 32 + lane] = (_Float16)0.f;
    }
  }
  __syncthreads();

  float* simW = simBuf + wave * 16 * NT;
  const int mtCount = B >> 4;

  // ---- Phase B: each wave independently processes M-tiles mt = wave, wave+8, ...
  for (int mt = wave; mt < mtCount; mt += NWAVES) {
    // A fragments for all 16 K-steps (16x32 f16 each), per documented layout:
    // lane: m = lane%16; VGPR v<4: K = k0 + 8*hf + 2v; v>=4: K = k0 + 16 + 8*hf + 2(v-4)
    Frag16 af[16];
    const _Float16* zrow = zn + (size_t)(mt * 16 + sub) * DDIM;
#pragma unroll
    for (int k = 0; k < 16; ++k) {
      const int k0 = k * 32 + hf * 8;
#pragma unroll
      for (int v = 0; v < 8; ++v) {
        const int kb = (v < 4) ? (k0 + 2 * v) : (k0 + 16 + 2 * (v - 4));
        af[k].u[v] = *(const uint32_t*)(zrow + kb);
      }
    }

    // 16 x NT sim tile, 16 columns at a time.
#pragma unroll 1
    for (int nt = 0; nt < NT / 16; ++nt) {
      const int nloc = nt * 16 + sub;
      v8f acc = {0.f, 0.f, 0.f, 0.f, 0.f, 0.f, 0.f, 0.f};
#pragma unroll
      for (int k = 0; k < 16; ++k) {
        // B fragment (32x16): lane: n = lane%16; VGPR v<4: K = 32k + 8*hf + 2v,+1; v>=4: K = 32k + 16 + 8*hf + 2(v-4),+1
        Frag16 bf;
        const _Float16* bp = tileB + nloc * BSTR + k * 32 + hf * 8;
#pragma unroll
        for (int v = 0; v < 8; ++v) bf.u[v] = *(const uint32_t*)(bp + ((v < 4) ? (2 * v) : (16 + 2 * (v - 4))));
        acc = __builtin_amdgcn_wmma_f32_16x16x32_f16(false, af[k].v, false, bf.v,
                                                     (short)0, acc, false, false);
      }
      // C/D layout: lane n = lane%16, VGPR r -> row m = r + 8*hf
#pragma unroll
      for (int r = 0; r < 8; ++r) simW[(r + 8 * hf) * NT + nloc] = acc[r];
    }
    __builtin_amdgcn_wave_barrier();  // keep LDS store->load order (intra-wave, DS in-order)

    // ---- chunk-local top-5 per M-row over NT columns (5 masked argmax passes)
    for (int row = 0; row < 16; ++row) {
      const float* srow = simW + row * NT;
      int   selC[TOPK];
      float selV[TOPK];
#pragma unroll
      for (int j = 0; j < TOPK; ++j) {
        float bv = -1e30f;
        int   bc = -1;
        for (int c = lane; c < NT; c += 32) {
          bool skip = (chunkBase + c >= N);
          for (int jj = 0; jj < j; ++jj) skip |= (c == selC[jj]);
          if (skip) continue;
          const float v = srow[c];
          if (v > bv) { bv = v; bc = c; }
        }
#pragma unroll
        for (int off = 16; off > 0; off >>= 1) {
          const float ov = __shfl_xor(bv, off, 32);
          const int   oc = __shfl_xor(bc, off, 32);
          if (ov > bv || (ov == bv && oc >= 0 && (bc < 0 || oc < bc))) { bv = ov; bc = oc; }
        }
        selC[j] = bc;
        selV[j] = bv;
      }
      if (lane < TOPK) {
        const int m = mt * 16 + row;
        const size_t base = ((size_t)blockIdx.x * B + m) * TOPK + lane;
        cand_val[base] = selV[lane];
        cand_idx[base] = (selC[lane] >= 0) ? (chunkBase + selC[lane]) : 0;
      }
    }
  }
}

// ---------------------------------------------------------------------------
// Kernel 3: merge per-chunk candidates -> global top-5, gather labels.
// One block per batch row b.
// ---------------------------------------------------------------------------
__global__ void __launch_bounds__(256)
reduce_topk(const float* __restrict__ cand_val, const int* __restrict__ cand_idx,
            const int* __restrict__ labels, float* __restrict__ out,
            int nchunks, int B) {
  __shared__ float sv[256 * TOPK];
  __shared__ int   si[256 * TOPK];
  const int b   = blockIdx.x;
  const int tid = threadIdx.x;
  const int total = nchunks * TOPK;

  float tv[TOPK];
  int   ti[TOPK];
#pragma unroll
  for (int j = 0; j < TOPK; ++j) { tv[j] = -1e30f; ti[j] = 0; }

  for (int c = tid; c < total; c += 256) {
    const int chunk = c / TOPK, j = c % TOPK;
    const size_t idx = ((size_t)chunk * B + b) * TOPK + j;
    const float v = cand_val[idx];
    if (v > tv[TOPK - 1]) {
      const int i = cand_idx[idx];
      int p = TOPK - 1;
      while (p > 0 && tv[p - 1] < v) { tv[p] = tv[p - 1]; ti[p] = ti[p - 1]; --p; }
      tv[p] = v; ti[p] = i;
    }
  }
#pragma unroll
  for (int j = 0; j < TOPK; ++j) { sv[tid * TOPK + j] = tv[j]; si[tid * TOPK + j] = ti[j]; }
  __syncthreads();

  if (tid < 32) {
#pragma unroll
    for (int j = 0; j < TOPK; ++j) { tv[j] = -1e30f; ti[j] = 0; }
    for (int t = tid; t < 256; t += 32) {
      for (int j = 0; j < TOPK; ++j) {
        const float v = sv[t * TOPK + j];
        if (v > tv[TOPK - 1]) {
          const int i = si[t * TOPK + j];
          int p = TOPK - 1;
          while (p > 0 && tv[p - 1] < v) { tv[p] = tv[p - 1]; ti[p] = ti[p - 1]; --p; }
          tv[p] = v; ti[p] = i;
        }
      }
    }
#pragma unroll
    for (int j = 0; j < TOPK; ++j) { sv[tid * TOPK + j] = tv[j]; si[tid * TOPK + j] = ti[j]; }
  }
  __syncthreads();

  if (tid == 0) {
    float fv[TOPK];
    int   fi[TOPK];
#pragma unroll
    for (int j = 0; j < TOPK; ++j) { fv[j] = -1e30f; fi[j] = 0; }
    for (int t = 0; t < 32; ++t) {
      for (int j = 0; j < TOPK; ++j) {
        const float v = sv[t * TOPK + j];
        if (v > fv[TOPK - 1]) {
          const int i = si[t * TOPK + j];
          int p = TOPK - 1;
          while (p > 0 && fv[p - 1] < v) { fv[p] = fv[p - 1]; fi[p] = fi[p - 1]; --p; }
          fv[p] = v; fi[p] = i;
        }
      }
    }
    // Output: [top_idx (B*5) | top_sim (B*5) | top_labels (B*5)] as floats.
    for (int j = 0; j < TOPK; ++j) {
      out[(size_t)b * TOPK + j]                         = (float)fi[j];
      out[(size_t)B * TOPK + (size_t)b * TOPK + j]      = fv[j];
      out[(size_t)2 * B * TOPK + (size_t)b * TOPK + j]  = (float)labels[fi[j]];
    }
  }
}

// ---------------------------------------------------------------------------
extern "C" void kernel_launch(void* const* d_in, const int* in_sizes, int n_in,
                              void* d_out, int out_size, void* d_ws, size_t ws_size,
                              hipStream_t stream) {
  (void)n_in; (void)out_size; (void)ws_size;
  const float* z      = (const float*)d_in[0];
  const float* bank   = (const float*)d_in[1];
  const int*   labels = (const int*)d_in[2];

  const int D = DDIM;
  const int B = in_sizes[0] / D;       // 1024
  const int N = in_sizes[2];           // 200000
  const int nchunks = (N + NT - 1) / NT;

  // Workspace layout: zn (f16, B*D) | cand_val (f32) | cand_idx (i32)
  _Float16* zn = (_Float16*)d_ws;
  size_t off = ((size_t)B * D * sizeof(_Float16) + 255) & ~(size_t)255;
  float* cand_val = (float*)((char*)d_ws + off);
  off += (size_t)nchunks * B * TOPK * sizeof(float);
  off = (off + 255) & ~(size_t)255;
  int* cand_idx = (int*)((char*)d_ws + off);

  norm_rows_f16<<<(B + NWAVES - 1) / NWAVES, 256, 0, stream>>>(z, zn, B);
  simtopk_kernel<<<nchunks, 256, 0, stream>>>(zn, bank, cand_val, cand_idx, N, B);
  reduce_topk<<<B, 256, 0, stream>>>(cand_val, cand_idx, labels, (float*)d_out, nchunks, B);
}
